// Head_84817014161731
// MI455X (gfx1250) — matrix operand order repaired and checked by reading, not compiled, not run on hardware
//
#include <hip/hip_runtime.h>

typedef __attribute__((ext_vector_type(16))) _Float16 v16h;
typedef __attribute__((ext_vector_type(8)))  float    v8f;
typedef unsigned int u32x4 __attribute__((ext_vector_type(4)));
typedef int          i32x8 __attribute__((ext_vector_type(8)));
typedef int          i32x4 __attribute__((ext_vector_type(4)));

namespace {
constexpr int kC  = 1024;   // model dim
constexpr int kH  = 64;     // head size
constexpr int kT  = 4096;   // sequence length
constexpr int kB  = 4;      // batch
}

// ---------------------------------------------------------------------------
// CDNA5 async / TDM helpers
// ---------------------------------------------------------------------------
__device__ inline void async_load_b128(unsigned int lds_addr, const void* gptr) {
  // LDS[lds_addr .. +16) = MEM[gptr .. +16), tracked with ASYNCcnt
  asm volatile("global_load_async_to_lds_b128 %0, %1, off"
               :: "v"(lds_addr), "v"((unsigned long long)(uintptr_t)gptr)
               : "memory");
}
__device__ inline void wait_asynccnt0() {
  asm volatile("s_wait_asynccnt 0x0" ::: "memory");
}

// 2D exact-fit TDM copy: tile_h rows of tile_w f16 elements (row stride
// `stride` elements) from global -> contiguous LDS.  D# per ISA 8.3-8.5.
__device__ inline void tdm_load_2d_f16(unsigned int lds_addr, const void* gptr,
                                       int tile_w, int tile_h, int stride) {
  const unsigned long long ga = (unsigned long long)(uintptr_t)gptr;
  u32x4 g0;
  g0[0] = 1u;                                   // count=1 (valid user D#)
  g0[1] = lds_addr;                             // lds_addr[31:0]
  g0[2] = (unsigned int)ga;                     // global_addr[31:0]
  g0[3] = (unsigned int)((ga >> 32) & 0x01FFFFFFu) | (2u << 30);  // type=2
  i32x8 g1;
  g1[0] = 1 << 16;                              // wg_mask=0, data_size=1 (2B)
  g1[1] = (tile_w & 0xFFFF) << 16;              // tensor_dim0[15:0]
  g1[2] = ((tile_w >> 16) & 0xFFFF) | ((tile_h & 0xFFFF) << 16);  // d0 hi | d1 lo
  g1[3] = ((tile_h >> 16) & 0xFFFF) | ((tile_w & 0xFFFF) << 16);  // d1 hi | tile_dim0
  g1[4] = tile_h & 0xFFFF;                      // tile_dim1, tile_dim2=0
  g1[5] = stride;                               // tensor_dim0_stride[31:0]
  g1[6] = 0;
  g1[7] = 0;
  i32x4 z4;
  z4[0] = 0; z4[1] = 0; z4[2] = 0; z4[3] = 0;
  i32x8 z8;
  z8[0] = 0; z8[1] = 0; z8[2] = 0; z8[3] = 0;
  z8[4] = 0; z8[5] = 0; z8[6] = 0; z8[7] = 0;
  // 6-arg toolchain variant: (g0, g1, g2, g3, g4, cpol)
  __builtin_amdgcn_tensor_load_to_lds(g0, g1, z4, z4, z8, 0);
}

// ---------------------------------------------------------------------------
// WMMA fragment loaders (CDNA5 16-bit layouts, wave32)
// ---------------------------------------------------------------------------
// A matrix 16x32 (MxK): lane = 16*half + m.
//   elems 0..7  -> K = 8*half + e ; elems 8..15 -> K = 16 + 8*half + e
template <typename ST>
__device__ inline v16h load_a_frag(const ST* base, int ld, int koff) {
  const int lane = threadIdx.x & 31;
  const int hs = lane >> 4;
  const int m  = lane & 15;
  const ST* p = base + (size_t)m * ld + koff + hs * 8;
  v16h a;
#pragma unroll
  for (int e = 0; e < 8; ++e) a[e] = (_Float16)p[e];
#pragma unroll
  for (int e = 0; e < 8; ++e) a[8 + e] = (_Float16)p[16 + e];
  return a;
}

// B matrix 32x16 (KxN): lane = 16*half + n; elems 0..7 -> K = 8*half + e ; elems 8..15 -> K = 16 + 8*half + e.
// `base` points at source[n=0][k=0] where source[n][k] == B[k][n].
__device__ inline v16h load_b_frag(const _Float16* base, int ld, int koff) {
  const int lane = threadIdx.x & 31;
  const int hs = lane >> 4;
  const int n  = lane & 15;
  const _Float16* p = base + (size_t)n * ld + koff + hs * 8;
  v16h b;
#pragma unroll
  for (int e = 0; e < 8; ++e) b[e] = p[e];
#pragma unroll
  for (int e = 0; e < 8; ++e) b[8 + e] = p[16 + e];
  return b;
}

__device__ inline v8f wmma_f16(v16h a, v16h b, v8f c) {
  return __builtin_amdgcn_wmma_f32_16x16x32_f16(false, a, false, b, (short)0, c,
                                                false, false);
}

// ---------------------------------------------------------------------------
// Kernel 1: weight-norm -> f16 weights.  w = g*v/||v||; 0.125 folded into Wq.
// ---------------------------------------------------------------------------
__global__ __launch_bounds__(256) void wnorm_kernel(
    const float* __restrict__ vq, const float* __restrict__ gq,
    const float* __restrict__ vk, const float* __restrict__ gk,
    const float* __restrict__ vv, const float* __restrict__ gv,
    _Float16* __restrict__ W) {
  const int which = blockIdx.x >> 6;
  const int row = blockIdx.x & 63;
  const float* v = (which == 0) ? vq : (which == 1) ? vk : vv;
  const float* g = (which == 0) ? gq : (which == 1) ? gk : gv;
  const float scale = (which == 0) ? 0.125f : 1.0f;
  const float* vr = v + (size_t)row * kC;

  float ss = 0.f;
  for (int c = threadIdx.x; c < kC; c += 256) { float t = vr[c]; ss += t * t; }
#pragma unroll
  for (int off = 16; off >= 1; off >>= 1) ss += __shfl_xor(ss, off, 32);

  __shared__ float red[8];
  __shared__ float coef;
  if ((threadIdx.x & 31) == 0) red[threadIdx.x >> 5] = ss;
  __syncthreads();
  if (threadIdx.x == 0) {
    float t = 0.f;
    for (int i = 0; i < 8; ++i) t += red[i];
    coef = g[row] * scale * rsqrtf(t);
  }
  __syncthreads();
  const float f = coef;
  _Float16* wr = W + (size_t)(which * kH + row) * kC;
  for (int c = threadIdx.x; c < kC; c += 256) wr[c] = (_Float16)(vr[c] * f);
}

// ---------------------------------------------------------------------------
// Kernel 2: fused QKV GEMM.  M=B*T, N=192, K=1024.
// W k-chunks double-buffered in LDS via async-to-LDS; prefetch overlaps WMMA.
// ---------------------------------------------------------------------------
__device__ inline void stage_w_chunk(const _Float16* __restrict__ W, int kc,
                                     unsigned int lds_base) {
  // chunk = W[0..192)[kc..kc+32) -> LDS row-major [192][32], 12KB, 768 x b128
  for (int i = threadIdx.x; i < 768; i += 128) {
    const int n = i >> 2;
    const int o = i & 3;
    async_load_b128(lds_base + (unsigned int)i * 16,
                    W + (size_t)n * kC + kc + o * 8);
  }
}

__global__ __launch_bounds__(128) void qkv_kernel(
    const float* __restrict__ x, const _Float16* __restrict__ W,
    _Float16* __restrict__ qb, _Float16* __restrict__ kb,
    _Float16* __restrict__ vb) {
  __shared__ __align__(16) _Float16 Wbuf[2][192 * 32];
  const int wave = threadIdx.x >> 5;
  const int m0 = blockIdx.x * 64 + wave * 16;
  const unsigned int wb0 = (unsigned int)(uintptr_t)&Wbuf[0][0];
  const unsigned int wb1 = (unsigned int)(uintptr_t)&Wbuf[1][0];

  v8f acc[12];
#pragma unroll
  for (int j = 0; j < 12; ++j)
#pragma unroll
    for (int i = 0; i < 8; ++i) acc[j][i] = 0.f;

  stage_w_chunk(W, 0, wb0);
  wait_asynccnt0();
  __syncthreads();

  for (int kc = 0; kc < kC; kc += 32) {
    const int cur = (kc >> 5) & 1;
    if (kc + 32 < kC) stage_w_chunk(W, kc + 32, cur ? wb0 : wb1);  // prefetch

    v16h a = load_a_frag(x + (size_t)m0 * kC, kC, kc);  // f32 -> f16
#pragma unroll
    for (int j = 0; j < 12; ++j) {
      v16h b = load_b_frag(&Wbuf[cur][j * 16 * 32], 32, 0);
      acc[j] = wmma_f16(a, b, acc[j]);
    }
    wait_asynccnt0();
    __syncthreads();
  }

  const int lane = threadIdx.x & 31;
  const int col = lane & 15, hs = lane >> 4;
#pragma unroll
  for (int j = 0; j < 12; ++j) {
    _Float16* dst = (j < 4) ? qb : (j < 8) ? kb : vb;
    const int nc = (j & 3) * 16 + col;
#pragma unroll
    for (int r = 0; r < 8; ++r) {
      const int row = m0 + r + 8 * hs;
      dst[(size_t)row * kH + nc] = (_Float16)acc[j][r];
    }
  }
}

// ---------------------------------------------------------------------------
// Kernel 3: causal flash attention.  Block = (batch, 64-query tile), 4 waves.
// K tile staged by the Tensor Data Mover; V staged transposed manually.
// ---------------------------------------------------------------------------
__global__ __launch_bounds__(128) void attn_kernel(
    const _Float16* __restrict__ qb, const _Float16* __restrict__ kb,
    const _Float16* __restrict__ vb, float* __restrict__ out) {
  const int b = blockIdx.y;
  const int q0 = blockIdx.x * 64;
  const int wave = threadIdx.x >> 5;
  const int lane = threadIdx.x & 31;
  const int col = lane & 15, hs = lane >> 4;
  const int qrow0 = q0 + wave * 16;

  __shared__ __align__(16) _Float16 Klds[64 * kH];   // K[k0+r][h], row-major
  __shared__ __align__(16) _Float16 Vt[kH][64];      // Vt[h][kk] = V[k0+kk][h]
  __shared__ __align__(16) _Float16 Plds[4][16][64]; // per-wave P tile
  const unsigned int klds_addr = (unsigned int)(uintptr_t)&Klds[0];

  const _Float16* qhead = qb + (size_t)(b * kT) * kH;
  const _Float16* khead = kb + (size_t)(b * kT) * kH;
  const _Float16* vhead = vb + (size_t)(b * kT) * kH;

  const v16h qa0 = load_a_frag(qhead + (size_t)qrow0 * kH, kH, 0);
  const v16h qa1 = load_a_frag(qhead + (size_t)qrow0 * kH, kH, 32);

  v8f o[4];
  float m_r[8], l_r[8];
#pragma unroll
  for (int f = 0; f < 4; ++f)
#pragma unroll
    for (int i = 0; i < 8; ++i) o[f][i] = 0.f;
#pragma unroll
  for (int r = 0; r < 8; ++r) { m_r[r] = -1e30f; l_r[r] = 0.f; }

  const int ntiles = q0 / 64 + 1;  // causal: key tiles 0 .. q0/64
  for (int kt = 0; kt < ntiles; ++kt) {
    const int k0 = kt * 64;

    __syncthreads();
    // TDM: DMA the 64x64 f16 K tile into LDS (wave 0 issues, TENSORcnt).
    if (threadIdx.x < 32)
      tdm_load_2d_f16(klds_addr, khead + (size_t)k0 * kH, kH, 64, kH);
    // All threads stage V transposed while the TDM runs.
    for (int i = threadIdx.x; i < 64 * kH; i += 128) {
      const int kk = i >> 6;
      const int h = i & 63;
      Vt[h][kk] = vhead[(size_t)(k0 + kk) * kH + h];
    }
    if (threadIdx.x < 32) __builtin_amdgcn_s_wait_tensorcnt(0);
    __syncthreads();

    // S = Q @ K^T   (B-columns of K^T are contiguous K rows in LDS)
    v8f s[4];
#pragma unroll
    for (int f = 0; f < 4; ++f) {
#pragma unroll
      for (int i = 0; i < 8; ++i) s[f][i] = 0.f;
      const _Float16* base = &Klds[(16 * f) * kH];
      s[f] = wmma_f16(qa0, load_b_frag(base, kH, 0), s[f]);
      s[f] = wmma_f16(qa1, load_b_frag(base, kH, 32), s[f]);
    }

    // Causal mask + online softmax in the f32 C-layout.
#pragma unroll
    for (int r = 0; r < 8; ++r) {
      const int row = qrow0 + r + 8 * hs;
      float mx = -1e30f;
#pragma unroll
      for (int f = 0; f < 4; ++f) {
        const int c = k0 + 16 * f + col;
        const float val = (c <= row) ? s[f][r] : -1e30f;
        s[f][r] = val;
        mx = fmaxf(mx, val);
      }
#pragma unroll
      for (int off = 1; off < 16; off <<= 1)
        mx = fmaxf(mx, __shfl_xor(mx, off, 32));
      const float mnew = fmaxf(m_r[r], mx);
      const float sc = __expf(m_r[r] - mnew);
      float rs = 0.f;
#pragma unroll
      for (int f = 0; f < 4; ++f) {
        const float p = __expf(s[f][r] - mnew);
        s[f][r] = p;
        rs += p;
      }
#pragma unroll
      for (int off = 1; off < 16; off <<= 1) rs += __shfl_xor(rs, off, 32);
      l_r[r] = l_r[r] * sc + rs;
      m_r[r] = mnew;
#pragma unroll
      for (int f = 0; f < 4; ++f) o[f][r] *= sc;
    }

    // C-layout -> A-layout via per-wave LDS round trip.
#pragma unroll
    for (int f = 0; f < 4; ++f)
#pragma unroll
      for (int r = 0; r < 8; ++r)
        Plds[wave][r + 8 * hs][16 * f + col] = (_Float16)s[f][r];
    asm volatile("s_wait_dscnt 0x0" ::: "memory");  // order cross-lane LDS RAW

    const v16h pa0 = load_a_frag(&Plds[wave][0][0], 64, 0);
    const v16h pa1 = load_a_frag(&Plds[wave][0][0], 64, 32);
#pragma unroll
    for (int f = 0; f < 4; ++f) {
      o[f] = wmma_f16(pa0, load_b_frag(&Vt[16 * f][0], 64, 0), o[f]);
      o[f] = wmma_f16(pa1, load_b_frag(&Vt[16 * f][0], 64, 32), o[f]);
    }
  }

  // Epilogue: normalize and store fp32.
#pragma unroll
  for (int f = 0; f < 4; ++f)
#pragma unroll
    for (int r = 0; r < 8; ++r) {
      const int row = qrow0 + r + 8 * hs;
      out[(size_t)(b * kT + row) * kH + 16 * f + col] = o[f][r] / l_r[r];
    }
}

// ---------------------------------------------------------------------------
extern "C" void kernel_launch(void* const* d_in, const int* in_sizes, int n_in,
                              void* d_out, int out_size, void* d_ws,
                              size_t ws_size, hipStream_t stream) {
  (void)in_sizes; (void)n_in; (void)out_size; (void)ws_size;
  const float* x  = (const float*)d_in[0];
  const float* vq = (const float*)d_in[1];
  const float* gq = (const float*)d_in[2];
  const float* vk = (const float*)d_in[3];
  const float* gk = (const float*)d_in[4];
  const float* vv = (const float*)d_in[5];
  const float* gv = (const float*)d_in[6];
  float* out = (float*)d_out;

  // Workspace (f16): W [192][1024], then q/k/v each [B*T][64].
  _Float16* W  = (_Float16*)d_ws;
  _Float16* qb = W + (size_t)192 * kC;
  _Float16* kb = qb + (size_t)kB * kT * kH;
  _Float16* vb = kb + (size_t)kB * kT * kH;

  wnorm_kernel<<<192, 256, 0, stream>>>(vq, gq, vk, gk, vv, gv, W);
  qkv_kernel<<<(kB * kT) / 64, 128, 0, stream>>>(x, W, qb, kb, vb);
  attn_kernel<<<dim3(kT / 64, kB), 128, 0, stream>>>(qb, kb, vb, out);
}
